// DGCNNSeg4D_1769526526593
// MI455X (gfx1250) — matrix operand order repaired and checked by reading, not compiled, not run on hardware
//
#include <hip/hip_runtime.h>

#define DEV __device__ __forceinline__

typedef __attribute__((ext_vector_type(16))) __bf16 v16bf;
typedef __attribute__((ext_vector_type(8)))  float  v8f;

union Frag { v16bf v; unsigned u[8]; };

DEV unsigned short f2bf(float f) {
  union { float f; unsigned u; } c{f};
  unsigned r = c.u + 0x7FFFu + ((c.u >> 16) & 1u);
  return (unsigned short)(r >> 16);
}
DEV float bf2f(unsigned short h) {
  union { unsigned u; float f; } c{((unsigned)h) << 16};
  return c.f;
}
DEV unsigned ld32(const unsigned short* p) { return *(const unsigned*)p; }

DEV v8f wmma_bf16(const Frag& a, const Frag& b, v8f c) {
  return __builtin_amdgcn_wmma_f32_16x16x32_bf16(
      false, a.v, false, b.v, (short)0, c, false, false);
}

// A fragment (16x32 bf16): lane = half*16+m, dword v -> k0 = (v>>2)*16+half*8+(v&3)*2
DEV void load_afrag(Frag& f, const unsigned short* row, int kb, int half) {
#pragma unroll
  for (int v = 0; v < 8; ++v)
    f.u[v] = ld32(row + kb + (v >> 2) * 16 + half * 8 + (v & 3) * 2);
}
// B fragment (32x16 bf16): lane = half*16+n, dword v -> k0 = (v>>2)*16+half*8+(v&3)*2
DEV void load_bfrag(Frag& f, const unsigned short* row, int kb, int half) {
#pragma unroll
  for (int v = 0; v < 8; ++v)
    f.u[v] = ld32(row + kb + (v >> 2) * 16 + half * 8 + (v & 3) * 2);
}

// ---------------------------------------------------------------------------
// Conversion kernels
// ---------------------------------------------------------------------------
__global__ void cvt_w_kernel(const float* __restrict__ w, int Ksrc, int koff,
                             int Kc, unsigned short* __restrict__ out, int Kp,
                             int total) {
  int i = blockIdx.x * 256 + threadIdx.x;
  if (i >= total) return;
  int o = i / Kp, k = i % Kp;
  out[i] = (k < Kc) ? f2bf(w[(size_t)o * Ksrc + koff + k]) : (unsigned short)0;
}

__global__ void cvt_x0_kernel(const float* __restrict__ x,
                              unsigned short* __restrict__ F, int total) {
  int i = blockIdx.x * 256 + threadIdx.x;
  if (i >= total) return;
  int r = i / 32, c = i % 32;
  F[i] = (c < 4) ? f2bf(x[(size_t)r * 4 + c]) : (unsigned short)0;
}

__global__ void prep_norm_kernel(const unsigned short* __restrict__ F, int ldf,
                                 int C, float* __restrict__ xx, int total) {
  int r = blockIdx.x * 256 + threadIdx.x;
  if (r >= total) return;
  const unsigned short* p = F + (size_t)r * ldf;
  float s = 0.f;
  for (int c = 0; c < C; ++c) { float f = bf2f(p[c]); s += f * f; }
  xx[r] = s;
}

__global__ void fill_zero_u32(unsigned* p, int n) {
  int i = blockIdx.x * 256 + threadIdx.x;
  if (i < n) p[i] = 0u;
}

// bias6[b][o] = sum_c w6[o][512+c] * xmax[b][c]
__global__ void bias6_kernel(const float* __restrict__ w6,
                             const unsigned* __restrict__ xmax,
                             float* __restrict__ bias6) {
  int i = blockIdx.x * 256 + threadIdx.x;
  if (i >= 4 * 512) return;
  int bb = i / 512, o = i % 512;
  const float* wr = w6 + (size_t)o * 1536 + 512;
  const unsigned* xm = xmax + bb * 1024;
  float s = 0.f;
  for (int c = 0; c < 1024; ++c) s += wr[c] * __uint_as_float(xm[c]);
  bias6[i] = s;
}

// ---------------------------------------------------------------------------
// kNN: Gram matrix via bf16 WMMA + register top-20 per query row.
// NCH = Cp/32 compile-time -> straight-line unrolled WMMA chunk sequence.
// ---------------------------------------------------------------------------
template <int NCH>
__global__ void knn_kernel(const unsigned short* __restrict__ F, int ldf,
                           const float* __restrict__ xx, int* __restrict__ idx,
                           int N) {
  __shared__ float dt[16][129];
  const int tid = threadIdx.x, lane = tid & 31, wv = tid >> 5;
  const int half = lane >> 4, n = lane & 15;
  const int b = blockIdx.x / (N / 16);
  const int q0 = (blockIdx.x % (N / 16)) * 16;
  const unsigned short* Fb = F + (size_t)b * N * ldf;
  const float* xxb = xx + (size_t)b * N;

  Frag aq[NCH];
  {
    const unsigned short* Fq = Fb + (size_t)(q0 + n) * ldf;
#pragma unroll
    for (int c = 0; c < NCH; ++c) load_afrag(aq[c], Fq, c * 32, half);
  }

  float bd[20];
  int bi[20];
#pragma unroll
  for (int s = 0; s < 20; ++s) { bd[s] = 3.4e38f; bi[s] = 0; }

  for (int cb = 0; cb < N; cb += 128) {
    const int c0 = cb + wv * 16;
    const unsigned short* Fc = Fb + (size_t)(c0 + n) * ldf;
    v8f acc = {};
#pragma unroll
    for (int c = 0; c < NCH; ++c) {
      Frag bf;
      load_bfrag(bf, Fc, c * 32, half);
      acc = wmma_bf16(aq[c], bf, acc);
    }
#pragma unroll
    for (int r = 0; r < 8; ++r) {
      int m = r + half * 8;
      dt[m][wv * 16 + n] = xxb[q0 + m] + xxb[c0 + n] - 2.f * acc[r];
    }
    __syncthreads();
    if (tid < 16) {
      for (int jj = 0; jj < 128; ++jj) {
        float d = dt[tid][jj];
        int j = cb + jj;
        float wmax = bd[0];
        int ws = 0;
#pragma unroll
        for (int s = 1; s < 20; ++s)
          if (bd[s] > wmax) { wmax = bd[s]; ws = s; }
        if (d < wmax) {
#pragma unroll
          for (int s = 0; s < 20; ++s)
            if (s == ws) { bd[s] = d; bi[s] = j; }
        }
      }
    }
    __syncthreads();
  }
  if (tid < 16) {
#pragma unroll
    for (int s = 0; s < 20; ++s)
      idx[((size_t)b * N + q0 + tid) * 20 + s] = bi[s];
  }
}

// ---------------------------------------------------------------------------
// EdgeConv: gather edge features ONCE into LDS-staged WMMA A fragments
// (per-lane-contiguous -> ds_store_b128/ds_load_b128), reuse over all O-tiles.
// 4 points/workgroup -> 80 rows = 5 M-tiles. NCH/OO compile-time: static LDS
// footprint (5*NCH KB) and fully static loop structure.
// ---------------------------------------------------------------------------
template <int NCH, int OO>
__global__ void edge_conv_kernel(const unsigned short* __restrict__ F, int ldf,
                                 int C, const int* __restrict__ idx,
                                 const unsigned short* __restrict__ W,
                                 const float* __restrict__ gw,
                                 const float* __restrict__ gb,
                                 const float* __restrict__ gm,
                                 const float* __restrict__ gv,
                                 unsigned short* __restrict__ out, int ldo,
                                 int N) {
  constexpr int Kp = NCH * 32;
  __shared__ unsigned abuf[5 * NCH * 256];  // [chunk*5+mtile][lane*8+v]
  __shared__ int idxl[80];
  __shared__ unsigned outb[4 * OO];
  const int tid = threadIdx.x, lane = tid & 31, wv = tid >> 5;
  const int half = lane >> 4, n = lane & 15;
  const int b = blockIdx.x / (N / 4);
  const int p0 = (blockIdx.x % (N / 4)) * 4;
  const unsigned short* Fb = F + (size_t)b * N * ldf;

  if (tid < 80)
    idxl[tid] = idx[((size_t)b * N + p0 + tid / 20) * 20 + tid % 20];
  for (int i = tid; i < 4 * OO; i += 256) outb[i] = 0u;
  __syncthreads();

  // ---- build phase: gather (nbr-ctr, ctr) into WMMA-A layout in LDS ----
  const int twoC = 2 * C;
  for (int bt = wv; bt < 5 * NCH; bt += 8) {
    const int mtile = bt % 5, chunk = bt / 5, kb = chunk * 32;
    const int Ra = mtile * 16 + n;
    const int gp = p0 + Ra / 20;
    const int nidx = idxl[Ra];
    unsigned* dst = &abuf[bt * 256 + lane * 8];
#pragma unroll
    for (int v = 0; v < 8; ++v) {
      int k = kb + (v >> 2) * 16 + half * 8 + (v & 3) * 2;
      unsigned pa;
      if (k < C) {  // neighbor - center
        unsigned un = ld32(Fb + (size_t)nidx * ldf + k);
        unsigned uc = ld32(Fb + (size_t)gp * ldf + k);
        float lo = bf2f((unsigned short)un) - bf2f((unsigned short)uc);
        float hi =
            bf2f((unsigned short)(un >> 16)) - bf2f((unsigned short)(uc >> 16));
        pa = (unsigned)f2bf(lo) | ((unsigned)f2bf(hi) << 16);
      } else if (k < twoC) {  // center
        pa = ld32(Fb + (size_t)gp * ldf + (k - C));
      } else {
        pa = 0u;  // K padding (layer 1)
      }
      dst[v] = pa;
    }
  }
  __syncthreads();

  // ---- compute phase: A from LDS, B (weights) from global (L2-resident) ----
  constexpr int total = 5 * (OO / 16);
  for (int t = wv; t < total; t += 8) {
    const int mtile = t % 5, otile = t / 5, ob = otile * 16;
    const unsigned short* Wr = W + (size_t)(ob + n) * Kp;
    v8f acc = {};
#pragma unroll
    for (int chunk = 0; chunk < NCH; ++chunk) {
      Frag a, bf;
      const unsigned* src = &abuf[(chunk * 5 + mtile) * 256 + lane * 8];
#pragma unroll
      for (int v = 0; v < 8; ++v) a.u[v] = src[v];
      load_bfrag(bf, Wr, chunk * 32, half);
      acc = wmma_bf16(a, bf, acc);
    }
    const int o = ob + n;
    float s = gw[o] * rsqrtf(gv[o] + 1e-5f);
    float bi = gb[o] - gm[o] * s;
#pragma unroll
    for (int r = 0; r < 8; ++r) {
      int R = mtile * 16 + r + half * 8;
      int p = R / 20;
      float h = fmaxf(acc[r] * s + bi, 0.f);
      atomicMax(&outb[p * OO + o], __float_as_uint(h));  // max over k (>=0)
    }
  }
  __syncthreads();
  for (int i = tid; i < 4 * OO; i += 256) {
    int p = i / OO, o = i % OO;
    out[((size_t)b * N + p0 + p) * ldo + o] = f2bf(__uint_as_float(outb[i]));
  }
}

// ---------------------------------------------------------------------------
// Generic bf16 WMMA GEMM: 32x32 tile per wave (2x2 WMMA sub-tiles),
// ping-pong software pipeline on K (K multiple of 64 for all launches).
// HAS1 compile-time selects 32-wide vs 16-wide O tile (layer 8: O=16).
// ---------------------------------------------------------------------------
template <bool HAS1>
__global__ void gemm_bf16_kernel(
    const unsigned short* __restrict__ A, int lda,
    const unsigned short* __restrict__ W, int ldw, int Mt2, int Ot2, int K,
    int O, const float* __restrict__ gw, const float* __restrict__ gb,
    const float* __restrict__ gm, const float* __restrict__ gv,
    const float* __restrict__ bbias, const float* __restrict__ rbias,
    unsigned short* __restrict__ outb, int ldob, float* __restrict__ outf,
    int ldof, unsigned* __restrict__ gmax, int NperB) {
  const int lane = threadIdx.x & 31, wv = threadIdx.x >> 5;
  const int half = lane >> 4, n = lane & 15;
  int t = blockIdx.x * 8 + wv;
  if (t >= Mt2 * Ot2) return;
  const int ot = t / Mt2, mt = t % Mt2;
  const int rb = mt * 32, ob = ot * 32;

  const unsigned short* Ar0 = A + (size_t)(rb + n) * lda;
  const unsigned short* Ar1 = A + (size_t)(rb + 16 + n) * lda;
  const unsigned short* Wr0 = W + (size_t)(ob + n) * ldw;
  const unsigned short* Wr1 = W + (size_t)(ob + (HAS1 ? 16 : 0) + n) * ldw;

  v8f acc00 = {}, acc01 = {}, acc10 = {}, acc11 = {};
  Frag a0A, a1A, b0A, b1A, a0B, a1B, b0B, b1B;
  load_afrag(a0A, Ar0, 0, half);
  load_afrag(a1A, Ar1, 0, half);
  load_bfrag(b0A, Wr0, 0, half);
  if (HAS1) load_bfrag(b1A, Wr1, 0, half);

  int kb = 0;
  while (true) {
    int kn = kb + 32;
    if (kn < K) {  // preload next chunk into B-set while A-set computes
      __builtin_prefetch(Ar0 + kn + 96, 0, 1);
      load_afrag(a0B, Ar0, kn, half);
      load_afrag(a1B, Ar1, kn, half);
      load_bfrag(b0B, Wr0, kn, half);
      if (HAS1) load_bfrag(b1B, Wr1, kn, half);
    }
    acc00 = wmma_bf16(a0A, b0A, acc00);
    acc10 = wmma_bf16(a1A, b0A, acc10);
    if (HAS1) {
      acc01 = wmma_bf16(a0A, b1A, acc01);
      acc11 = wmma_bf16(a1A, b1A, acc11);
    }
    kb = kn;
    if (kb >= K) break;

    kn = kb + 32;
    if (kn < K) {  // preload into A-set while B-set computes
      load_afrag(a0A, Ar0, kn, half);
      load_afrag(a1A, Ar1, kn, half);
      load_bfrag(b0A, Wr0, kn, half);
      if (HAS1) load_bfrag(b1A, Wr1, kn, half);
    }
    acc00 = wmma_bf16(a0B, b0B, acc00);
    acc10 = wmma_bf16(a1B, b0B, acc10);
    if (HAS1) {
      acc01 = wmma_bf16(a0B, b1B, acc01);
      acc11 = wmma_bf16(a1B, b1B, acc11);
    }
    kb = kn;
    if (kb >= K) break;
  }

  const int o0 = ob + n;
  const int o1 = HAS1 ? ob + 16 + n : o0;
  float s0 = 1.f, bi0 = 0.f, s1 = 1.f, bi1 = 0.f;
  if (gw) {
    s0 = gw[o0] * rsqrtf(gv[o0] + 1e-5f);
    bi0 = gb[o0] - gm[o0] * s0;
    if (HAS1) {
      s1 = gw[o1] * rsqrtf(gv[o1] + 1e-5f);
      bi1 = gb[o1] - gm[o1] * s1;
    }
  }
  if (rbias) { bi0 += rbias[o0]; if (HAS1) bi1 += rbias[o1]; }

  auto emit = [&](float h, int row, int o, float s, float bi) {
    if (bbias) h += bbias[(row / NperB) * O + o];
    h = h * s + bi;
    if (gw) h = fmaxf(h, 0.f);
    if (gmax) atomicMax(&gmax[(row / NperB) * O + o], __float_as_uint(h));
    if (outb) outb[(size_t)row * ldob + o] = f2bf(h);
    if (outf) outf[(size_t)row * ldof + o] = h;
  };
#pragma unroll
  for (int r = 0; r < 8; ++r) {
    int row0 = rb + r + half * 8, row1 = row0 + 16;
    emit(acc00[r], row0, o0, s0, bi0);
    emit(acc10[r], row1, o0, s0, bi0);
    if (HAS1) {
      emit(acc01[r], row0, o1, s1, bi1);
      emit(acc11[r], row1, o1, s1, bi1);
    }
  }
}

// ---------------------------------------------------------------------------
extern "C" void kernel_launch(void* const* d_in, const int* in_sizes, int n_in,
                              void* d_out, int out_size, void* d_ws,
                              size_t ws_size, hipStream_t stream) {
  (void)in_sizes; (void)n_in; (void)out_size; (void)ws_size;
  const int B = 4, N = 4096, NB = B * N;

  const float* x = (const float*)d_in[0];
  const float *w[8], *gg[8], *gb_[8], *gm[8], *gv[8];
  for (int i = 1; i <= 7; ++i) {
    int base = 1 + (i - 1) * 5;
    w[i]   = (const float*)d_in[base];
    gg[i]  = (const float*)d_in[base + 1];
    gb_[i] = (const float*)d_in[base + 2];
    gm[i]  = (const float*)d_in[base + 3];
    gv[i]  = (const float*)d_in[base + 4];
  }
  const float* w8v = (const float*)d_in[36];
  const float* b8v = (const float*)d_in[37];

  char* ws = (char*)d_ws;
  size_t off = 0;
  auto alloc = [&](size_t bytes) -> char* {
    char* p = ws + off;
    off = (off + bytes + 255) & ~(size_t)255;
    return p;
  };

  unsigned short* feat0 = (unsigned short*)alloc((size_t)NB * 32 * 2);
  unsigned short* xcat  = (unsigned short*)alloc((size_t)NB * 512 * 2);
  unsigned short* f6    = (unsigned short*)alloc((size_t)NB * 512 * 2);
  unsigned short* f7    = (unsigned short*)alloc((size_t)NB * 256 * 2);
  float* xx             = (float*)alloc((size_t)NB * 4);
  int* idx              = (int*)alloc((size_t)NB * 20 * 4);
  unsigned short* wb1   = (unsigned short*)alloc(64 * 32 * 2);
  unsigned short* wb2   = (unsigned short*)alloc(64 * 128 * 2);
  unsigned short* wb3   = (unsigned short*)alloc(128 * 128 * 2);
  unsigned short* wb4   = (unsigned short*)alloc(256 * 256 * 2);
  unsigned short* wb5   = (unsigned short*)alloc(1024 * 512 * 2);
  unsigned short* wb6   = (unsigned short*)alloc(512 * 512 * 2);
  unsigned short* wb7   = (unsigned short*)alloc(256 * 512 * 2);
  unsigned short* wb8   = (unsigned short*)alloc(16 * 256 * 2);
  unsigned* xmax        = (unsigned*)alloc(4 * 1024 * 4);
  float* bias6          = (float*)alloc(4 * 512 * 4);

  auto gridN = [](int total) { return (total + 255) / 256; };

  cvt_w_kernel<<<gridN(64 * 32), 256, 0, stream>>>(w[1], 8, 0, 8, wb1, 32, 64 * 32);
  cvt_w_kernel<<<gridN(64 * 128), 256, 0, stream>>>(w[2], 128, 0, 128, wb2, 128, 64 * 128);
  cvt_w_kernel<<<gridN(128 * 128), 256, 0, stream>>>(w[3], 128, 0, 128, wb3, 128, 128 * 128);
  cvt_w_kernel<<<gridN(256 * 256), 256, 0, stream>>>(w[4], 256, 0, 256, wb4, 256, 256 * 256);
  cvt_w_kernel<<<gridN(1024 * 512), 256, 0, stream>>>(w[5], 512, 0, 512, wb5, 512, 1024 * 512);
  cvt_w_kernel<<<gridN(512 * 512), 256, 0, stream>>>(w[6], 1536, 0, 512, wb6, 512, 512 * 512);
  cvt_w_kernel<<<gridN(256 * 512), 256, 0, stream>>>(w[7], 512, 0, 512, wb7, 512, 256 * 512);
  cvt_w_kernel<<<gridN(16 * 256), 256, 0, stream>>>(w8v, 256, 0, 256, wb8, 256, 16 * 256);

  cvt_x0_kernel<<<gridN(NB * 32), 256, 0, stream>>>(x, feat0, NB * 32);

  const int knnBlocks = B * (N / 16);
  const int ecBlocks = B * (N / 4);

  // ---- EdgeConv 1 (C=4 -> O=64, Kp=32) ----
  prep_norm_kernel<<<gridN(NB), 256, 0, stream>>>(feat0, 32, 4, xx, NB);
  knn_kernel<1><<<knnBlocks, 256, 0, stream>>>(feat0, 32, xx, idx, N);
  edge_conv_kernel<1, 64><<<ecBlocks, 256, 0, stream>>>(
      feat0, 32, 4, idx, wb1, gg[1], gb_[1], gm[1], gv[1], xcat + 0, 512, N);
  // ---- EdgeConv 2 (C=64 -> O=64, Kp=128) ----
  prep_norm_kernel<<<gridN(NB), 256, 0, stream>>>(xcat + 0, 512, 64, xx, NB);
  knn_kernel<2><<<knnBlocks, 256, 0, stream>>>(xcat + 0, 512, xx, idx, N);
  edge_conv_kernel<4, 64><<<ecBlocks, 256, 0, stream>>>(
      xcat + 0, 512, 64, idx, wb2, gg[2], gb_[2], gm[2], gv[2], xcat + 64, 512, N);
  // ---- EdgeConv 3 (C=64 -> O=128, Kp=128) ----
  prep_norm_kernel<<<gridN(NB), 256, 0, stream>>>(xcat + 64, 512, 64, xx, NB);
  knn_kernel<2><<<knnBlocks, 256, 0, stream>>>(xcat + 64, 512, xx, idx, N);
  edge_conv_kernel<4, 128><<<ecBlocks, 256, 0, stream>>>(
      xcat + 64, 512, 64, idx, wb3, gg[3], gb_[3], gm[3], gv[3], xcat + 128, 512, N);
  // ---- EdgeConv 4 (C=128 -> O=256, Kp=256) ----
  prep_norm_kernel<<<gridN(NB), 256, 0, stream>>>(xcat + 128, 512, 128, xx, NB);
  knn_kernel<4><<<knnBlocks, 256, 0, stream>>>(xcat + 128, 512, xx, idx, N);
  edge_conv_kernel<8, 256><<<ecBlocks, 256, 0, stream>>>(
      xcat + 128, 512, 128, idx, wb4, gg[4], gb_[4], gm[4], gv[4], xcat + 256, 512, N);

  const int Mt2 = NB / 32;  // 512
  // ---- layer 5: GEMM K=512 O=1024, BN+ReLU, global max only ----
  fill_zero_u32<<<gridN(4 * 1024), 256, 0, stream>>>(xmax, 4 * 1024);
  {
    int tiles = Mt2 * (1024 / 32);
    gemm_bf16_kernel<true><<<(tiles + 7) / 8, 256, 0, stream>>>(
        xcat, 512, wb5, 512, Mt2, 1024 / 32, 512, 1024, gg[5], gb_[5], gm[5],
        gv[5], nullptr, nullptr, nullptr, 0, nullptr, 0, xmax, N);
  }
  // ---- fold broadcast global-max branch of layer 6 into per-batch bias ----
  bias6_kernel<<<gridN(4 * 512), 256, 0, stream>>>(w[6], xmax, bias6);
  // ---- layer 6: GEMM K=512 O=512 + bias6, BN+ReLU -> f6 (bf16) ----
  {
    int tiles = Mt2 * (512 / 32);
    gemm_bf16_kernel<true><<<(tiles + 7) / 8, 256, 0, stream>>>(
        xcat, 512, wb6, 512, Mt2, 512 / 32, 512, 512, gg[6], gb_[6], gm[6],
        gv[6], bias6, nullptr, f6, 512, nullptr, 0, nullptr, N);
  }
  // ---- layer 7: GEMM K=512 O=256, BN+ReLU -> f7 (bf16) ----
  {
    int tiles = Mt2 * (256 / 32);
    gemm_bf16_kernel<true><<<(tiles + 7) / 8, 256, 0, stream>>>(
        f6, 512, wb7, 512, Mt2, 256 / 32, 512, 256, gg[7], gb_[7], gm[7],
        gv[7], nullptr, nullptr, f7, 256, nullptr, 0, nullptr, N);
  }
  // ---- layer 8: GEMM K=256 O=16 + b8 -> d_out (f32, (B,N,16)) ----
  {
    int tiles = Mt2 * 1;
    gemm_bf16_kernel<false><<<(tiles + 7) / 8, 256, 0, stream>>>(
        f7, 256, wb8, 256, Mt2, 1, 256, 16, nullptr, nullptr, nullptr, nullptr,
        nullptr, b8v, nullptr, 0, (float*)d_out, 16, nullptr, N);
  }
}
